// AttentionRNNCell_38096359916084
// MI455X (gfx1250) — matrix operand order repaired and checked by reading, not compiled, not run on hardware
//
#include <hip/hip_runtime.h>
#include <hip/hip_bf16.h>

// ---------------------------------------------------------------------------
// CDNA5 WMMA types (wave32): v_wmma_f32_16x16x32_bf16
// ---------------------------------------------------------------------------
typedef __attribute__((ext_vector_type(16))) __bf16          v16bf;
typedef __attribute__((ext_vector_type(8)))  float           v8f;
typedef __attribute__((ext_vector_type(8)))  unsigned short  ushort8;

#define BATCH 512
#define HDIM  400
#define FOURH 1600
#define CDIM  73
#define XD    3
#define UDIM  2048
#define KMIX  10
#define KSTR  896          // padded K stride (max Ktot=876 -> 896, mult of 32)
#define NSLICE 8           // attention u-slices per batch element

__device__ __forceinline__ unsigned short f2bf_bits(float f) {
    unsigned u = __float_as_uint(f);
    unsigned r = u + 0x7FFFu + ((u >> 16) & 1u);   // round-to-nearest-even
    return (unsigned short)(r >> 16);
}

union FragBF { ushort8 h[2]; v16bf v; };

// Virtual concat: A = [seg0 | seg1 | seg2 | hseg], each [B x len] row-major.
__device__ __forceinline__ float fetchA(int m, int k,
                                        const float* s0, int n0,
                                        const float* s1, int n1,
                                        const float* s2, int n2,
                                        const float* hs) {
    if (k < n0) return s0[m * n0 + k];
    k -= n0;
    if (k < n1) return s1[m * n1 + k];
    k -= n1;
    if (n2 > 0 && k < n2) return s2[m * n2 + k];
    if (n2 > 0) k -= n2;
    return hs[m * HDIM + k];
}

// ---------------------------------------------------------------------------
// Pack A panel to bf16: Apack[m][k], k zero-padded to Kpad. One thread = 8 k's.
// ---------------------------------------------------------------------------
__global__ void pack_a_bf16(const float* s0, int n0, const float* s1, int n1,
                            const float* s2, int n2, const float* hseg,
                            unsigned short* __restrict__ Apack,
                            int Ktot, int Kpad) {
    const int kg8 = Kpad >> 3;
    const int tid = blockIdx.x * blockDim.x + threadIdx.x;
    if (tid >= BATCH * kg8) return;
    const int m  = tid / kg8;
    const int kg = (tid % kg8) << 3;
    ushort8 v;
    #pragma unroll
    for (int j = 0; j < 8; ++j) {
        const int k = kg + j;
        const float f = (k < Ktot) ? fetchA(m, k, s0, n0, s1, n1, s2, n2, hseg) : 0.0f;
        v[j] = f2bf_bits(f);
    }
    *(ushort8*)(Apack + (long)m * KSTR + kg) = v;
}

// ---------------------------------------------------------------------------
// Pack weights [Wx ; Ur] (Ktot x 1600) to bf16, TRANSPOSED: WpackT[n][k].
// Consecutive threads -> consecutive n => coalesced f32 reads.
// ---------------------------------------------------------------------------
__global__ void pack_w_bf16(const float* __restrict__ Wx, const float* __restrict__ Ur,
                            unsigned short* __restrict__ WpackT,
                            int Kin, int Ktot, int Kpad) {
    const int tid = blockIdx.x * blockDim.x + threadIdx.x;
    const int kg8 = Kpad >> 3;
    if (tid >= FOURH * kg8) return;
    const int kg = (tid / FOURH) << 3;
    const int n  = tid % FOURH;
    ushort8 v;
    #pragma unroll
    for (int j = 0; j < 8; ++j) {
        const int k = kg + j;
        float f = 0.0f;
        if (k < Ktot)
            f = (k < Kin) ? Wx[(long)k * FOURH + n] : Ur[(long)(k - Kin) * FOURH + n];
        v[j] = f2bf_bits(f);
    }
    *(ushort8*)(WpackT + (long)n * KSTR + kg) = v;
}

// ---------------------------------------------------------------------------
// Z = Apack @ WpackT^T  -> (512 x 1600) f32.
// One wave owns a 16(M) x 64(N) tile (4 accumulators). No LDS, no branches:
// A fragment = two 16B K-contiguous loads (reused by 4 WMMAs);
// B fragment = two 16B K-contiguous loads per 16-wide subtile.
// jobs = 32 Mtiles x 25 Ntiles = 800 waves = 200 blocks of 4 waves.
// ---------------------------------------------------------------------------
__global__ void __launch_bounds__(128)
lstm_gemm_wmma(const unsigned short* __restrict__ Apack,
               const unsigned short* __restrict__ WpackT,
               float* __restrict__ z, int Kpad) {
    const int wave = threadIdx.x >> 5;
    const int lane = threadIdx.x & 31;
    const int half = lane >> 4;
    const int l16  = lane & 15;

    const int job   = blockIdx.x * 4 + wave;
    const int mtile = (job / 25) * 16;
    const int nbase = (job % 25) * 64;

    const unsigned short* Arow = Apack + (long)(mtile + l16) * KSTR;
    const unsigned short* Brow0 = WpackT + (long)(nbase + l16) * KSTR + half * 16;

    v8f acc0 = {}, acc1 = {}, acc2 = {}, acc3 = {};

    for (int k0 = 0; k0 < Kpad; k0 += 32) {
        FragBF a;
        // lane holds A[M=l16][K]: K-contiguous run {half*16 .. +15}
        a.h[0] = *(const ushort8*)(Arow + k0 + half * 16);
        a.h[1] = *(const ushort8*)(Arow + k0 + half * 16 + 8);

        FragBF b;
        // lane holds B[K][N=l16]: K-contiguous run {half*16 .. +15} (WpackT layout)
        const unsigned short* bp = Brow0 + k0;
        b.h[0] = *(const ushort8*)(bp);
        b.h[1] = *(const ushort8*)(bp + 8);
        acc0 = __builtin_amdgcn_wmma_f32_16x16x32_bf16(false, a.v, false, b.v, (short)0, acc0, false, false);

        bp += 16 * KSTR;
        b.h[0] = *(const ushort8*)(bp);
        b.h[1] = *(const ushort8*)(bp + 8);
        acc1 = __builtin_amdgcn_wmma_f32_16x16x32_bf16(false, a.v, false, b.v, (short)0, acc1, false, false);

        bp += 16 * KSTR;
        b.h[0] = *(const ushort8*)(bp);
        b.h[1] = *(const ushort8*)(bp + 8);
        acc2 = __builtin_amdgcn_wmma_f32_16x16x32_bf16(false, a.v, false, b.v, (short)0, acc2, false, false);

        bp += 16 * KSTR;
        b.h[0] = *(const ushort8*)(bp);
        b.h[1] = *(const ushort8*)(bp + 8);
        acc3 = __builtin_amdgcn_wmma_f32_16x16x32_bf16(false, a.v, false, b.v, (short)0, acc3, false, false);
    }

    // C/D layout: VGPR r -> M = half*8 + r, N = lane%16 (+ 16*subtile)
    float* zr = z + (long)(mtile + half * 8) * FOURH + nbase + l16;
    #pragma unroll
    for (int r = 0; r < 8; ++r) {
        zr[(long)r * FOURH +  0] = acc0[r];
        zr[(long)r * FOURH + 16] = acc1[r];
        zr[(long)r * FOURH + 32] = acc2[r];
        zr[(long)r * FOURH + 48] = acc3[r];
    }
}

// ---------------------------------------------------------------------------
// Peephole LSTM gate nonlinearity (elementwise epilogue; adds bias here).
// ---------------------------------------------------------------------------
__global__ void __launch_bounds__(256)
lstm_gates(const float* __restrict__ z, const float* __restrict__ bias,
           const float* __restrict__ peep, const float* __restrict__ c_prev,
           float* __restrict__ h_out, float* __restrict__ c_out,
           float* __restrict__ h_out2) {
    const int idx = blockIdx.x * blockDim.x + threadIdx.x;
    if (idx >= BATCH * HDIM) return;
    const int b = idx / HDIM;
    const int h = idx % HDIM;
    const float* zr = z + (long)b * FOURH;

    const float c  = c_prev[idx];
    const float zi = zr[h]            + bias[h];
    const float zf = zr[HDIM + h]     + bias[HDIM + h];
    const float zg = zr[2 * HDIM + h] + bias[2 * HDIM + h];
    const float zo = zr[3 * HDIM + h] + bias[3 * HDIM + h];

    const float ig = 1.0f / (1.0f + __expf(-(zi + c * peep[h])));
    const float fg = 1.0f / (1.0f + __expf(-(zf + c * peep[HDIM + h])));
    const float cn = fg * c + ig * tanhf(zg);
    const float og = 1.0f / (1.0f + __expf(-(zo + cn * peep[2 * HDIM + h])));
    const float hn = og * tanhf(cn);

    h_out[idx] = hn;
    c_out[idx] = cn;
    if (h_out2) h_out2[idx] = hn;
}

// ---------------------------------------------------------------------------
// Attention window. grid = (B, NSLICE), block = 128.
// Streams the 306 MB one-hot tensor with coalesced 73-lane rows + prefetch.
// ---------------------------------------------------------------------------
__global__ void __launch_bounds__(128)
attention_kernel(const float* __restrict__ w_prev, const float* __restrict__ x,
                 const float* __restrict__ s1h,
                 const float* __restrict__ Wa, const float* __restrict__ ba,
                 const float* __restrict__ kappa, const int* __restrict__ seqlen,
                 const float* __restrict__ onehot,
                 float* __restrict__ kappa_out, float* __restrict__ w_out) {
    const int b     = blockIdx.x;
    const int slice = blockIdx.y;                 // 0..NSLICE-1
    const int tid   = threadIdx.x;
    const int USL   = UDIM / NSLICE;              // u's per slice

    __shared__ float alpha[KMIX], beta[KMIX], knew[KMIX];
    __shared__ float phi[128];

    if (tid < 3 * KMIX) {
        float a = ba[tid];
        for (int i = 0; i < CDIM; ++i) a += w_prev[b * CDIM + i] * Wa[i * 3 * KMIX + tid];
        for (int i = 0; i < XD;   ++i) a += x[b * XD + i]        * Wa[(CDIM + i) * 3 * KMIX + tid];
        for (int i = 0; i < HDIM; ++i) a += s1h[b * HDIM + i]    * Wa[(CDIM + XD + i) * 3 * KMIX + tid];
        const float e = __expf(a);
        if (tid < KMIX)            alpha[tid] = e;
        else if (tid < 2 * KMIX)   beta[tid - KMIX] = e;
        else {
            const float kn = kappa[b * KMIX + (tid - 2 * KMIX)] + e;
            knew[tid - 2 * KMIX] = kn;
            if (slice == 0) kappa_out[b * KMIX + (tid - 2 * KMIX)] = kn;
        }
    }
    __syncthreads();

    const int len = seqlen[b];
    const int u0  = slice * USL;
    float acc = 0.0f;                             // thread tid owns channel c = tid (< 73)

    for (int uc = 0; uc < USL; uc += 128) {
        const int u = u0 + uc + tid;
        float ph = 0.0f;
        if (u < len) {
            #pragma unroll
            for (int k = 0; k < KMIX; ++k) {
                const float d = knew[k] - (float)u;
                ph += alpha[k] * __expf(-beta[k] * d * d);
            }
        }
        phi[tid] = ph;

        // prefetch next chunk of one-hot rows (-> global_prefetch_b8)
        if (uc + 128 < USL) {
            const char* nxt = (const char*)(onehot + ((long)b * UDIM + u0 + uc + 128) * CDIM);
            __builtin_prefetch(nxt + tid * 292, 0, 1);
        }
        __syncthreads();

        if (tid < CDIM) {
            const float* oh = onehot + ((long)b * UDIM + u0 + uc) * CDIM + tid;
            for (int uu = 0; uu < 128; ++uu) {
                const float p = phi[uu];          // uniform across lanes
                if (p != 0.0f) acc += p * oh[(long)uu * CDIM];
            }
        }
        __syncthreads();
    }
    if (tid < CDIM) atomicAdd(&w_out[b * CDIM + tid], acc);
}

__global__ void zero_kernel(float* p, int n) {
    const int i = blockIdx.x * blockDim.x + threadIdx.x;
    if (i < n) p[i] = 0.0f;
}

// ---------------------------------------------------------------------------
// Host-side launch sequence (stream-ordered dependency chain).
// ---------------------------------------------------------------------------
extern "C" void kernel_launch(void* const* d_in, const int* in_sizes, int n_in,
                              void* d_out, int out_size, void* d_ws, size_t ws_size,
                              hipStream_t stream) {
    const float* x      = (const float*)d_in[0];
    const float* s1_h   = (const float*)d_in[1];
    const float* s1_c   = (const float*)d_in[2];
    const float* s2_h   = (const float*)d_in[3];
    const float* s2_c   = (const float*)d_in[4];
    const float* s3_h   = (const float*)d_in[5];
    const float* s3_c   = (const float*)d_in[6];
    const float* kappa  = (const float*)d_in[7];
    const float* w_prev = (const float*)d_in[8];
    const float* onehot = (const float*)d_in[9];
    const int*   slen   = (const int*)  d_in[10];
    const float* W1 = (const float*)d_in[11];
    const float* U1 = (const float*)d_in[12];
    const float* b1 = (const float*)d_in[13];
    const float* p1 = (const float*)d_in[14];
    const float* W2 = (const float*)d_in[15];
    const float* U2 = (const float*)d_in[16];
    const float* b2 = (const float*)d_in[17];
    const float* p2 = (const float*)d_in[18];
    const float* W3 = (const float*)d_in[19];
    const float* U3 = (const float*)d_in[20];
    const float* b3 = (const float*)d_in[21];
    const float* p3 = (const float*)d_in[22];
    const float* Wa = (const float*)d_in[23];
    const float* ba = (const float*)d_in[24];

    float* out = (float*)d_out;
    const int BH = BATCH * HDIM;                 // 204800
    float* o_s3h   = out;                        // return[0]
    float* o_s1h   = out + 1 * BH;               // return[1]
    float* o_s1c   = out + 2 * BH;               // return[2]
    float* o_s2h   = out + 3 * BH;               // return[3]
    float* o_s2c   = out + 4 * BH;               // return[4]
    float* o_s3h2  = out + 5 * BH;               // return[5] (dup of s3_h_new)
    float* o_s3c   = out + 6 * BH;               // return[6]
    float* o_kappa = out + 7 * BH;               // return[7]  B*K
    float* o_w     = out + 7 * BH + BATCH * KMIX;// return[8]  B*C

    // workspace carve-up (16B-aligned offsets)
    char* ws = (char*)d_ws;
    float*          z      = (float*)         (ws);                 // 512*1600*4 = 3,276,800
    unsigned short* Apack  = (unsigned short*)(ws + 3276800);       // 512*896*2  =   917,504
    unsigned short* WpackT = (unsigned short*)(ws + 3276800 + 917504); // 1600*896*2 = 2,867,200

    const int K1tot = CDIM + XD + HDIM;          // 476
    const int K1pad = 480;
    const int K2tot = XD + HDIM + CDIM + HDIM;   // 876
    const int K2pad = 896;

    const int gemm_blocks = (32 * 25) / 4;       // 800 wave-jobs / 4 waves = 200
    const int gate_blocks = (BATCH * HDIM + 255) / 256;
    const int pa1 = (BATCH * (K1pad >> 3) + 255) / 256;
    const int pa2 = (BATCH * (K2pad >> 3) + 255) / 256;
    const int pw1 = (FOURH * (K1pad >> 3) + 255) / 256;
    const int pw2 = (FOURH * (K2pad >> 3) + 255) / 256;

    // zero the atomic accumulation target for w_new (d_out is poisoned)
    zero_kernel<<<(BATCH * CDIM + 255) / 256, 256, 0, stream>>>(o_w, BATCH * CDIM);

    // ---- layer 1: z = [w_prev|x] @ W1 + s1_h @ U1 ----
    pack_a_bf16<<<pa1, 256, 0, stream>>>(w_prev, CDIM, x, XD, nullptr, 0, s1_h,
                                         Apack, K1tot, K1pad);
    pack_w_bf16<<<pw1, 256, 0, stream>>>(W1, U1, WpackT, CDIM + XD, K1tot, K1pad);
    lstm_gemm_wmma<<<gemm_blocks, 128, 0, stream>>>(Apack, WpackT, z, K1pad);
    lstm_gates<<<gate_blocks, 256, 0, stream>>>(z, b1, p1, s1_c, o_s1h, o_s1c, nullptr);

    // ---- attention window (streams the 306 MB one-hot tensor) ----
    attention_kernel<<<dim3(BATCH, NSLICE), 128, 0, stream>>>(
        w_prev, x, o_s1h, Wa, ba, kappa, slen, onehot, o_kappa, o_w);

    // ---- layer 2: z = [x|s1_h_new|w_new] @ W2 + s2_h @ U2 ----
    pack_a_bf16<<<pa2, 256, 0, stream>>>(x, XD, o_s1h, HDIM, o_w, CDIM, s2_h,
                                         Apack, K2tot, K2pad);
    pack_w_bf16<<<pw2, 256, 0, stream>>>(W2, U2, WpackT, XD + HDIM + CDIM, K2tot, K2pad);
    lstm_gemm_wmma<<<gemm_blocks, 128, 0, stream>>>(Apack, WpackT, z, K2pad);
    lstm_gates<<<gate_blocks, 256, 0, stream>>>(z, b2, p2, s2_c, o_s2h, o_s2c, nullptr);

    // ---- layer 3: z = [x|s2_h_new|w_new] @ W3 + s3_h @ U3 ----
    pack_a_bf16<<<pa2, 256, 0, stream>>>(x, XD, o_s2h, HDIM, o_w, CDIM, s3_h,
                                         Apack, K2tot, K2pad);
    pack_w_bf16<<<pw2, 256, 0, stream>>>(W3, U3, WpackT, XD + HDIM + CDIM, K2tot, K2pad);
    lstm_gemm_wmma<<<gemm_blocks, 128, 0, stream>>>(Apack, WpackT, z, K2pad);
    lstm_gates<<<gate_blocks, 256, 0, stream>>>(z, b3, p3, s3_c, o_s3h, o_s3c, o_s3h2);
}
